// KW_CascadedBranchPlus_24936580120849
// MI455X (gfx1250) — matrix operand order repaired and checked by reading, not compiled, not run on hardware
//
#include <hip/hip_runtime.h>
#include <math.h>

#define BN     512     // B*N rows of kw
#define DMODEL 768
#define TDIM   512
#define VOCAB  49408
#define VT     32               // emb rows (columns of score) per block
#define NBLK   (VOCAB / VT)     // 1544 score blocks

typedef __attribute__((ext_vector_type(16))) __bf16 v16bf;
typedef __attribute__((ext_vector_type(8)))  float  v8f;

// ---- order-preserving float<->u32 mapping for packed argmax ----
__device__ __forceinline__ unsigned f32_ord(float f) {
  unsigned b = __float_as_uint(f);
  return (b & 0x80000000u) ? ~b : (b | 0x80000000u);
}
__device__ __forceinline__ float ord_f32(unsigned u) {
  unsigned b = (u & 0x80000000u) ? (u ^ 0x80000000u) : ~u;
  return __uint_as_float(b);
}
__device__ __forceinline__ unsigned long long shfl_xor_u64(unsigned long long v, int m) {
  unsigned lo = __shfl_xor((unsigned)(v & 0xFFFFFFFFull), m, 32);
  unsigned hi = __shfl_xor((unsigned)(v >> 32), m, 32);
  return ((unsigned long long)hi << 32) | lo;
}

// Swizzle (m,k) -> flat index in the pre-packed A-fragment array.
// A-fragment (16x32 bf16): lanes 0-15 hold M rows with K 0..15 contiguous;
// lanes 16-31 hold K 16..31 (same contiguous order as the B fragments).
__device__ __forceinline__ int kwA_index(int m, int k) {
  int m_tile  = m >> 4, lane_lo = m & 15;
  int kstep   = k >> 5, kk = k & 31;
  int g       = kk >> 4;          // half-wave: K 0..15 / 16..31
  int e       = kk & 15;          // element index within lane
  int lane    = (g << 4) | lane_lo;
  return ((m_tile * 16 + kstep) * (32 * 16)) + lane * 16 + e;
}

// ------------------- kernel 1: projection + L2 norm + bf16 swizzle -------------------
__global__ void __launch_bounds__(256)
proj_norm_kernel(const float* __restrict__ audio, const float* __restrict__ W,
                 const float* __restrict__ bias, __bf16* __restrict__ kwA) {
  __shared__ float arow[DMODEL];
  __shared__ float red[256];
  const int tid = threadIdx.x;
  const int m   = blockIdx.x;

  for (int i = tid; i < DMODEL; i += 256) arow[i] = audio[(size_t)m * DMODEL + i];
  __syncthreads();

  float acc0 = bias[tid], acc1 = bias[tid + 256];
  for (int k = 0; k < DMODEL; ++k) {
    float a = arow[k];
    acc0 = fmaf(a, W[(size_t)k * TDIM + tid],       acc0);
    acc1 = fmaf(a, W[(size_t)k * TDIM + tid + 256], acc1);
  }

  red[tid] = acc0 * acc0 + acc1 * acc1;
  __syncthreads();
  for (int s = 128; s > 0; s >>= 1) {
    if (tid < s) red[tid] += red[tid + s];
    __syncthreads();
  }
  const float inv = 1.0f / fmaxf(sqrtf(red[0]), 1e-8f);

  kwA[kwA_index(m, tid)]       = (__bf16)(acc0 * inv);
  kwA[kwA_index(m, tid + 256)] = (__bf16)(acc1 * inv);
}

// ------------------- kernel 2: fused WMMA score GEMM + online argmax/sumexp -------------------
__global__ void __launch_bounds__(256)
score_kernel(const __bf16* __restrict__ kwA, const float* __restrict__ emb,
             unsigned long long* __restrict__ gpack, float* __restrict__ gsum) {
  __shared__ __bf16 ldsB[VT][TDIM];               // 32 KB: emb tile, bf16
  __shared__ float  s_rinv[VT];
  __shared__ unsigned long long s_pack[BN][2];    // 8 KB: per-row {score,arg} per n-tile
  __shared__ float  s_sum[BN][2];                 // 4 KB: per-row sumexp per n-tile

  const int tid   = threadIdx.x;
  const int vbase = blockIdx.x * VT;

  // stage emb tile -> bf16 LDS (coalesced float4 reads)
  const float4* src = (const float4*)(emb + (size_t)vbase * TDIM);
  for (int i = tid; i < VT * TDIM / 4; i += 256) {
    float4 v = src[i];
    int row = i >> 7;            // TDIM/4 = 128 float4 per row
    int k   = (i & 127) << 2;
    ldsB[row][k]     = (__bf16)v.x;
    ldsB[row][k + 1] = (__bf16)v.y;
    ldsB[row][k + 2] = (__bf16)v.z;
    ldsB[row][k + 3] = (__bf16)v.w;
  }
  __syncthreads();

  if (tid < VT) {   // per emb row: 1/max(||emb||, eps)
    float s = 0.f;
    for (int k = 0; k < TDIM; ++k) { float x = (float)ldsB[tid][k]; s = fmaf(x, x, s); }
    s_rinv[tid] = 1.0f / fmaxf(sqrtf(s), 1e-8f);
  }
  __syncthreads();

  const int wave   = tid >> 5, lane = tid & 31;
  const int ntile  = wave & 1;          // which 16-column tile
  const int q      = wave >> 1;         // which quarter of the 32 m-tiles
  const int nlocal = ntile * 16 + (lane & 15);
  const int g      = lane >> 4;
  const float rinvv = s_rinv[nlocal];
  const int vcol   = vbase + nlocal;

  // Preload all 16 B fragments into registers (each lane: 16 contiguous
  // K values of emb row `nlocal`, same k order as the packed kwA A fragments).
  v16bf bfr[16];
#pragma unroll
  for (int ks = 0; ks < 16; ++ks)
    bfr[ks] = *(const v16bf*)&ldsB[nlocal][ks * 32 + g * 16];

  const __bf16* aBase = kwA + lane * 16;

  // Two m-tiles per iteration: two independent accumulator chains
  // + A-fragment double buffering so loads overlap the WMMAs.
  for (int mi = 0; mi < 8; mi += 2) {
    const int mt0 = q * 8 + mi;
    const int mt1 = mt0 + 1;
    v8f acc0 = {}, acc1 = {};

    v16bf a0 = *(const v16bf*)(aBase + (mt0 * 16 + 0) * 512);
    v16bf a1 = *(const v16bf*)(aBase + (mt1 * 16 + 0) * 512);
#pragma unroll
    for (int ks = 0; ks < 16; ++ks) {
      v16bf a0n = a0, a1n = a1;
      if (ks < 15) {   // prefetch next step's A fragments before this step's WMMAs
        a0n = *(const v16bf*)(aBase + (mt0 * 16 + ks + 1) * 512);
        a1n = *(const v16bf*)(aBase + (mt1 * 16 + ks + 1) * 512);
      }
      acc0 = __builtin_amdgcn_wmma_f32_16x16x32_bf16(
               false, a0, false, bfr[ks], (short)0, acc0, false, false);
      acc1 = __builtin_amdgcn_wmma_f32_16x16x32_bf16(
               false, a1, false, bfr[ks], (short)0, acc1, false, false);
      a0 = a0n; a1 = a1n;
    }

    // epilogue: VGPR r holds row m = mtile*16 + g*8 + r, column = lane&15
#pragma unroll
    for (int c = 0; c < 2; ++c) {
      const v8f acc   = c ? acc1 : acc0;
      const int mtile = c ? mt1  : mt0;
      for (int r = 0; r < 8; ++r) {
        float s = acc[r] * rinvv;                       // cosine score
        float e = __expf(s * 10.0f);                    // exp(s / tau)
        unsigned long long pk =
          ((unsigned long long)f32_ord(s) << 32) | (unsigned)(~(unsigned)vcol);
        for (int msk = 1; msk < 16; msk <<= 1) {        // reduce over 16 columns
          e += __shfl_xor(e, msk, 32);
          unsigned long long o = shfl_xor_u64(pk, msk);
          pk = (o > pk) ? o : pk;
        }
        if ((lane & 15) == 0) {
          int row = mtile * 16 + g * 8 + r;
          s_pack[row][ntile] = pk;
          s_sum[row][ntile]  = e;
        }
      }
    }
  }
  __syncthreads();

  // combine the two n-tiles, write partials coalesced for the reducer
  for (int row = tid; row < BN; row += 256) {
    unsigned long long p0 = s_pack[row][0], p1 = s_pack[row][1];
    gpack[(size_t)row * NBLK + blockIdx.x] = (p1 > p0) ? p1 : p0;
    gsum [(size_t)row * NBLK + blockIdx.x] = s_sum[row][0] + s_sum[row][1];
  }
}

// ------------------- kernel 3: cross-tile reduce + straight-through gather -------------------
__global__ void __launch_bounds__(256)
reduce_gather_kernel(const unsigned long long* __restrict__ gpack,
                     const float* __restrict__ gsum,
                     const float* __restrict__ emb,
                     float* __restrict__ out) {
  __shared__ unsigned long long rpk[256];
  __shared__ float rsm[256];
  __shared__ float s_factor;
  __shared__ unsigned s_v;

  const int tid = threadIdx.x;
  const int row = blockIdx.x;

  unsigned long long best = 0ull;
  float ssum = 0.f;
  for (int i = tid; i < NBLK; i += 256) {
    unsigned long long p = gpack[(size_t)row * NBLK + i];
    best = (p > best) ? p : best;
    ssum += gsum[(size_t)row * NBLK + i];
  }
  rpk[tid] = best; rsm[tid] = ssum;
  __syncthreads();
  for (int s = 128; s > 0; s >>= 1) {
    if (tid < s) {
      if (rpk[tid + s] > rpk[tid]) rpk[tid] = rpk[tid + s];
      rsm[tid] += rsm[tid + s];
    }
    __syncthreads();
  }
  if (tid == 0) {
    unsigned long long b = rpk[0];
    float smax   = ord_f32((unsigned)(b >> 32));
    unsigned vst = ~(unsigned)(b & 0xFFFFFFFFull);
    float p      = __expf(smax * 10.0f) / rsm[0];   // softmax prob at argmax
    s_factor = (1.0f + p) - p;                      // straight-through residual
    s_v      = vst;
  }
  __syncthreads();
  const unsigned v = s_v;
  const float    f = s_factor;
  for (int t = tid; t < TDIM; t += 256)
    out[(size_t)row * TDIM + t] = emb[(size_t)v * TDIM + t] * f;
}

extern "C" void kernel_launch(void* const* d_in, const int* in_sizes, int n_in,
                              void* d_out, int out_size, void* d_ws, size_t ws_size,
                              hipStream_t stream) {
  (void)in_sizes; (void)n_in; (void)out_size; (void)ws_size;
  const float* audio = (const float*)d_in[0];   // [8,64,768]
  const float* W     = (const float*)d_in[1];   // [768,512]
  const float* bias  = (const float*)d_in[2];   // [512]
  const float* emb   = (const float*)d_in[3];   // [49408,512]
  float* out = (float*)d_out;                   // [8,64,512]

  char* w = (char*)d_ws;
  __bf16* kwA = (__bf16*)w;                                     // 512 KB
  unsigned long long* gpack =
      (unsigned long long*)(w + (size_t)BN * TDIM * 2);         // 512*1544*8
  float* gsum =
      (float*)(w + (size_t)BN * TDIM * 2 + (size_t)BN * NBLK * 8);

  proj_norm_kernel<<<BN, 256, 0, stream>>>(audio, W, bias, kwA);
  score_kernel<<<NBLK, 256, 0, stream>>>(kwA, emb, gpack, gsum);
  reduce_gather_kernel<<<BN, 256, 0, stream>>>(gpack, gsum, emb, out);
}
